// Regression_17377437680524
// MI455X (gfx1250) — matrix operand order repaired and checked by reading, not compiled, not run on hardware
//
#include <hip/hip_runtime.h>
#include <hip/hip_bf16.h>

// ---------------------------------------------------------------------------
// Types for CDNA5 WMMA
// ---------------------------------------------------------------------------
typedef __attribute__((ext_vector_type(16))) __bf16 v16bf;
typedef __attribute__((ext_vector_type(8)))  __bf16 v8bf;
typedef __attribute__((ext_vector_type(8)))  float  v8f;

union V16U { v16bf v; v8bf h[2]; };

#define LOG2PI_F 1.8378770664093453f
#define IN_F 2048

__device__ __forceinline__ unsigned short f2bf(float f) {
    union { float f; unsigned int u; } v; v.f = f;
    unsigned int u = v.u;
    unsigned int r = u + 0x7FFFu + ((u >> 16) & 1u);   // round-to-nearest-even
    return (unsigned short)(r >> 16);
}

// ---------------------------------------------------------------------------
// Kernel 1: build LT (bf16, transposed L) via 64x64 LDS tile transpose and
// accumulate sum(L*L) in f32 from the source (f32) values.
//   L[k][j] = (k>j) ? W_offdiag[k][j] : (k==j ? exp(W_logdiag[k]) : 0)
//   LT[j*IN + k] = bf16(L[k][j])
// Grid: 32x32 tiles = 1024 blocks x 256 threads. Reads and writes coalesced.
// ---------------------------------------------------------------------------
#define TPAD 72   // 64 + 8 ushort padding

__global__ __launch_bounds__(256)
void k_build_lt(const float* __restrict__ Woff, const float* __restrict__ Wlog,
                unsigned short* __restrict__ LT, float* __restrict__ acc)
{
    __shared__ unsigned short tile[64 * TPAD];
    __shared__ float red[256];

    const int t  = threadIdx.x;
    const int tj = blockIdx.x & 31;    // 64-col (j) tile
    const int tk = blockIdx.x >> 5;    // 64-row (k) tile

    // ---- read phase: coalesced over j (row k fixed per 64-thread group)
    float ss = 0.0f;
    const int lc = t & 63;             // local j
    #pragma unroll
    for (int rep = 0; rep < 16; ++rep) {
        const int lr = rep * 4 + (t >> 6);        // local k, 0..63
        const int k  = tk * 64 + lr;
        const int j  = tj * 64 + lc;
        float val = 0.0f;
        if (k > j)       val = Woff[(size_t)k * IN_F + j];
        else if (k == j) val = __expf(Wlog[k]);
        ss += val * val;
        tile[lr * TPAD + lc] = f2bf(val);
    }
    __syncthreads();

    // ---- write phase: coalesced over k (row j fixed per 64-thread group)
    const int kc = t & 63;             // local k
    #pragma unroll
    for (int rep = 0; rep < 16; ++rep) {
        const int jr = rep * 4 + (t >> 6);        // local j, 0..63
        LT[(size_t)(tj * 64 + jr) * IN_F + tk * 64 + kc] = tile[kc * TPAD + jr];
    }

    // ---- reduce sum(L^2)
    red[t] = ss;
    __syncthreads();
    for (int s = 128; s > 0; s >>= 1) {
        if (t < s) red[t] += red[t + s];
        __syncthreads();
    }
    if (t == 0) unsafeAtomicAdd(&acc[2], red[0]);
}

// ---------------------------------------------------------------------------
// Kernel 2: fused  Y = X @ L ; quad[b] = ||Y[b]||^2 ; pred[b] = x[b].W_mean ;
//           per-row ELBO contribution accumulated atomically.
// One WG = 128 threads = 4 waves; owns 32 rows of X (full K=2048, bf16 in
// LDS).  Wave w covers j-blocks [w*32, w*32+32); each wave computes TWO
// 16x16 C tiles (row-tiles 0-15 and 16-31) sharing every B load, halving the
// L2-resident LT stream.  Simple loop body (loads declared inside, moderate
// unroll) so the compiler software-pipelines WITHOUT spilling to scratch.
// ---------------------------------------------------------------------------
#define XPAD 2056   // 2048 + 8 bf16 pad -> row stride 4112B, de-conflicts LDS banks

__global__ __launch_bounds__(128)
void k_main(const float* __restrict__ x, const float* __restrict__ y,
            const float* __restrict__ counter,
            const float* __restrict__ noise_logdiag,
            const float* __restrict__ W_mean,
            const unsigned short* __restrict__ LT,
            float* __restrict__ acc)
{
    __shared__ __align__(16) unsigned short ldsX[32 * XPAD]; // 131584 B
    __shared__ float predRow[32];
    __shared__ float wavePartial[128];
    __shared__ float redc[32], redn[32];

    const int t    = threadIdx.x;
    const int lane = t & 31;
    const int w    = t >> 5;
    const int b0   = blockIdx.x * 32;

    if (t < 32) predRow[t] = 0.0f;
    __syncthreads();

    // ---- Phase 1: load 32x2048 f32 tile -> bf16 LDS, dot with W_mean
    const float4* X4  = (const float4*)x;
    const float4* WM4 = (const float4*)W_mean;
    for (int r = 0; r < 32; ++r) {
        float dp = 0.0f;
        #pragma unroll
        for (int q = 0; q < 4; ++q) {
            const int c4 = t + 128 * q;                 // 0..511
            float4 xv = X4[(size_t)(b0 + r) * 512 + c4];
            float4 wv = WM4[c4];
            dp += xv.x * wv.x + xv.y * wv.y + xv.z * wv.z + xv.w * wv.w;
            unsigned int u0 = (unsigned int)f2bf(xv.x) |
                              ((unsigned int)f2bf(xv.y) << 16);
            unsigned int u1 = (unsigned int)f2bf(xv.z) |
                              ((unsigned int)f2bf(xv.w) << 16);
            *(uint2*)&ldsX[r * XPAD + c4 * 4] = make_uint2(u0, u1);
        }
        atomicAdd(&predRow[r], dp);     // ds_add_f32
    }
    __syncthreads();

    // ---- Phase 2: WMMA over this wave's 32 j-blocks, full K, 2 M-tiles
    const int m     = lane & 15;                 // A row id / B-C column id
    const int base0 = (lane < 16) ? 0 : 8;       // A K-group offset
    const int kbase = (lane < 16) ? 0 : 16;      // unused; B loads also use base0
    const unsigned short* arow0 = &ldsX[m * XPAD];
    const unsigned short* arow1 = &ldsX[(m + 16) * XPAD];

    float sq0[8], sq1[8];
    #pragma unroll
    for (int i = 0; i < 8; ++i) { sq0[i] = 0.0f; sq1[i] = 0.0f; }

    for (int jb = w * 32; jb < w * 32 + 32; ++jb) {
        const unsigned short* brow = LT + (size_t)(jb * 16 + m) * IN_F;
        v8f c0 = {}, c1 = {};
        #pragma unroll 4
        for (int k0 = 0; k0 < IN_F; k0 += 32) {
            V16U a0, a1, bb;
            a0.h[0] = *(const v8bf*)&arow0[k0 + base0];
            a0.h[1] = *(const v8bf*)&arow0[k0 + 16 + base0];
            a1.h[0] = *(const v8bf*)&arow1[k0 + base0];
            a1.h[1] = *(const v8bf*)&arow1[k0 + 16 + base0];
            bb.h[0] = *(const v8bf*)&brow[k0 + base0];
            bb.h[1] = *(const v8bf*)&brow[k0 + 16 + base0];
            c0 = __builtin_amdgcn_wmma_f32_16x16x32_bf16(
                     false, a0.v, false, bb.v, (short)0, c0, false, false);
            c1 = __builtin_amdgcn_wmma_f32_16x16x32_bf16(
                     false, a1.v, false, bb.v, (short)0, c1, false, false);
        }
        #pragma unroll
        for (int i = 0; i < 8; ++i) { sq0[i] += c0[i] * c0[i];
                                      sq1[i] += c1[i] * c1[i]; }
    }

    // ---- Phase 3: reduce squares across the 16 lanes of each half-wave
    #pragma unroll
    for (int mask = 8; mask >= 1; mask >>= 1) {
        #pragma unroll
        for (int i = 0; i < 8; ++i) {
            sq0[i] += __shfl_xor(sq0[i], mask, 32);
            sq1[i] += __shfl_xor(sq1[i], mask, 32);
        }
    }
    if (lane == 0) {
        #pragma unroll
        for (int i = 0; i < 8; ++i) {
            wavePartial[w * 32 + i]      = sq0[i];   // rows 0..7
            wavePartial[w * 32 + 16 + i] = sq1[i];   // rows 16..23
        }
    }
    if (lane == 16) {
        #pragma unroll
        for (int i = 0; i < 8; ++i) {
            wavePartial[w * 32 + 8 + i]  = sq0[i];   // rows 8..15
            wavePartial[w * 32 + 24 + i] = sq1[i];   // rows 24..31
        }
    }
    __syncthreads();

    // ---- Final per-row ELBO contribution
    if (t < 32) {
        const float quad = wavePartial[t] + wavePartial[32 + t] +
                           wavePartial[64 + t] + wavePartial[96 + t];
        const float pred = predRow[t];
        const int   b    = b0 + t;
        const float c    = counter[b];
        const float nld  = noise_logdiag[0];
        const float inv  = __expf(-nld);
        const float dz   = (y[b] - pred) * inv;
        const float plik = -0.5f * dz * dz - nld - 0.5f * LOG2PI_F;
        const float tr   = 0.5f * quad * c * c * inv * inv;
        redc[t] = (plik - tr) * c;
        redn[t] = c;
    }
    __syncthreads();
    if (t == 0) {
        float s0 = 0.0f, s1 = 0.0f;
        #pragma unroll
        for (int i = 0; i < 32; ++i) { s0 += redc[i]; s1 += redn[i]; }
        unsafeAtomicAdd(&acc[0], s0);
        unsafeAtomicAdd(&acc[1], s1);
    }
}

// ---------------------------------------------------------------------------
// Kernel 3: finalize scalar terms, write -elbo.
// ---------------------------------------------------------------------------
__global__ __launch_bounds__(256)
void k_final(const float* __restrict__ W_mean, const float* __restrict__ Wlog,
             const float* __restrict__ noise_logdiag,
             const float* __restrict__ acc, float* __restrict__ out)
{
    __shared__ float rm[256], rl[256];
    float mse = 0.0f, slog = 0.0f;
    for (int i = threadIdx.x; i < IN_F; i += 256) {
        float wv = W_mean[i];
        mse  += wv * wv;
        slog += Wlog[i];
    }
    rm[threadIdx.x] = mse; rl[threadIdx.x] = slog;
    __syncthreads();
    for (int s = 128; s > 0; s >>= 1) {
        if (threadIdx.x < s) {
            rm[threadIdx.x] += rm[threadIdx.x + s];
            rl[threadIdx.x] += rl[threadIdx.x + s];
        }
        __syncthreads();
    }
    if (threadIdx.x == 0) {
        const float nld      = noise_logdiag[0];
        const float data     = acc[0] / acc[1];
        const float traceCov = acc[2];
        // KL(q||N(0,I)): prior_scale = 1 -> log term 0
        const float kl    = 0.5f * (rm[0] + traceCov - 2.0f * rl[0]);
        const float nprec = __expf(-2.0f * nld);
        // adjusted_dof = (1 + 1 + 1)/2 = 1.5 ; wishart_scale = 0.01
        const float wish  = 1.5f * (-2.0f * nld) - 0.5f * 0.01f * nprec;
        const float elbo  = data + 1.0e-4f * (wish - kl);
        out[0] = -elbo;
    }
}

// ---------------------------------------------------------------------------
// Launch
// ---------------------------------------------------------------------------
extern "C" void kernel_launch(void* const* d_in, const int* in_sizes, int n_in,
                              void* d_out, int out_size, void* d_ws, size_t ws_size,
                              hipStream_t stream) {
    const float* x            = (const float*)d_in[0];   // (16384,2048)
    const float* y            = (const float*)d_in[1];   // (16384,1)
    const float* counter      = (const float*)d_in[2];   // (16384,)
    // d_in[3] = noise_mean (unused by reference)
    const float* noise_logdiag= (const float*)d_in[4];   // (1,)
    const float* W_mean       = (const float*)d_in[5];   // (1,2048)
    const float* W_logdiag    = (const float*)d_in[6];   // (1,2048)
    const float* W_offdiag    = (const float*)d_in[7];   // (1,2048,2048)

    float*          acc = (float*)d_ws;                               // acc[0..2]
    unsigned short* LT  = (unsigned short*)((char*)d_ws + 256);       // 8 MB bf16

    hipMemsetAsync(d_ws, 0, 3 * sizeof(float), stream);

    k_build_lt<<<1024, 256, 0, stream>>>(W_offdiag, W_logdiag, LT, acc);

    k_main<<<16384 / 32, 128, 0, stream>>>(x, y, counter, noise_logdiag,
                                           W_mean, LT, acc);

    k_final<<<1, 256, 0, stream>>>(W_mean, W_logdiag, noise_logdiag,
                                   acc, (float*)d_out);
}
